// Model_2619930051467
// MI455X (gfx1250) — matrix operand order repaired and checked by reading, not compiled, not run on hardware
//
#include <hip/hip_runtime.h>
#include <hip/hip_bf16.h>

// ---------------------------------------------------------------------------
// Model dims (compile-time constants from the reference)
// ---------------------------------------------------------------------------
#define L_LAYERS 16
#define HD 64
#define NH 15
#define NKV 5
#define TH 960
#define TI 2560
#define EH 720
#define EI 2048
#define PL 512
#define EL 128
#define SEQ (PL + EL)   // 640
#define DQ (NH * HD)    // 960
#define DKV (NKV * HD)  // 320

typedef __attribute__((ext_vector_type(16))) __bf16 v16bf;
typedef __attribute__((ext_vector_type(8)))  __bf16 v8bf;
typedef __attribute__((ext_vector_type(8)))  float  v8f;

// Convert 8 fp32 -> 8 bf16 and store 16 contiguous bytes to LDS (ds_store_b128)
__device__ __forceinline__ void bf16_store8(void* dst, const float* s) {
  v8bf v;
  #pragma unroll
  for (int i = 0; i < 8; ++i) v[i] = (__bf16)s[i];
  *(v8bf*)dst = v;
}

// ---------------------------------------------------------------------------
// Tiled GEMM: C[M,N] = A[M,K] @ B[K,N] (+ R if R != null), fp32 row-major.
// Block tile 128x64, 256 threads = 8 waves, each wave owns a 32x32 sub-tile
// (2x2 grid of 16x16 accumulators -> 4 WMMAs per K-step, K-step = 32).
// Software-pipelined: next K-step's A/B tiles are prefetched into registers
// right after the barrier, so global_load latency overlaps the WMMAs.
// A/B tiles live in LDS in *fragment order* (lane's 16 bf16 = 32 contiguous
// bytes), so each fragment is a single 32-byte LDS read. Layouts per CDNA5
// ISA 7.12.2 (kg = k/8):
//   A (16x32): element (row16,k): lane = row16 + ((kg&1)<<4), half = kg>>1
//   B (32x16): element (k,col16): lane = col16 + ((kg&1)<<4), half = kg>>1
// ---------------------------------------------------------------------------
__global__ __launch_bounds__(256)
void gemm_bf16_wmma(const float* __restrict__ A, const float* __restrict__ B,
                    float* __restrict__ C, const float* __restrict__ R,
                    int M, int N, int K) {
  // A: 8 sub-tiles (16 rows each) x 1024 B; B: 4 sub-tiles (16 cols) x 1024 B
  __shared__ __align__(32) unsigned char Alds[8 * 1024];
  __shared__ __align__(32) unsigned char Blds[4 * 1024];

  const int tid  = threadIdx.x;
  const int wave = tid >> 5;        // 0..7
  const int lane = tid & 31;
  const int mb = blockIdx.y * 128;
  const int nb = blockIdx.x * 64;
  const int wr = (wave >> 1) * 32;  // wave row base within tile (0/32/64/96)
  const int wc = (wave & 1) * 32;   // wave col base within tile (0/32)
  const int row16 = lane & 15;

  char* Abase = (char*)Alds;
  char* Bbase = (char*)Blds;
  const char* aptr0 = Abase + ((wr >> 4) + 0) * 1024 + lane * 32;
  const char* aptr1 = Abase + ((wr >> 4) + 1) * 1024 + lane * 32;
  const char* bptr0 = Bbase + ((wc >> 4) + 0) * 1024 + lane * 32;
  const char* bptr1 = Bbase + ((wc >> 4) + 1) * 1024 + lane * 32;

  // Staging destinations (fixed per thread)
  // A: slot sl -> r = sl>>2 (0..127), kg = sl&3 ; two slots: tid, tid+256
  // B: kg = tid>>6 (0..3), n = tid&63
  const int ar0 = tid >> 2,           akg0 = tid & 3;
  const int ar1 = (tid + 256) >> 2,   akg1 = tid & 3;   // +256 keeps kg
  char* adst0 = Abase + (ar0 >> 4) * 1024 +
                ((ar0 & 15) + ((akg0 & 1) << 4)) * 32 + (akg0 >> 1) * 16;
  char* adst1 = Abase + (ar1 >> 4) * 1024 +
                ((ar1 & 15) + ((akg1 & 1) << 4)) * 32 + (akg1 >> 1) * 16;
  const int bkg = tid >> 6;
  const int bn  = tid & 63;
  char* bdst = Bbase + (bn >> 4) * 1024 +
               ((bn & 15) + ((bkg & 1) << 4)) * 32 + (bkg >> 1) * 16;

  v8f c00 = {}, c01 = {}, c10 = {}, c11 = {};

  float arg[2][8];   // prefetched A slots
  float brg[8];      // prefetched B slot

  auto load_tiles = [&](int kk) {
    const bool fk = (kk + 32 <= K);
    const bool fA = fk && (mb + 128 <= M);
    const bool fB = fk && (nb + 64 <= N);
    const int rr[2]  = { ar0, ar1 };
    const int kgg[2] = { akg0, akg1 };
    #pragma unroll
    for (int it = 0; it < 2; ++it) {
      if (fA) {
        const float4* src =
            (const float4*)(A + (size_t)(mb + rr[it]) * K + (kk + kgg[it] * 8));
        const float4 p0 = src[0];
        const float4 p1 = src[1];
        arg[it][0] = p0.x; arg[it][1] = p0.y; arg[it][2] = p0.z; arg[it][3] = p0.w;
        arg[it][4] = p1.x; arg[it][5] = p1.y; arg[it][6] = p1.z; arg[it][7] = p1.w;
      } else {
        const int gr = mb + rr[it];
        #pragma unroll
        for (int i = 0; i < 8; ++i) {
          const int gc = kk + kgg[it] * 8 + i;
          arg[it][i] = (gr < M && gc < K) ? A[(size_t)gr * K + gc] : 0.0f;
        }
      }
    }
    if (fB) {
      #pragma unroll
      for (int i = 0; i < 8; ++i)
        brg[i] = B[(size_t)(kk + bkg * 8 + i) * N + (nb + bn)];
    } else {
      #pragma unroll
      for (int i = 0; i < 8; ++i) {
        const int gk = kk + bkg * 8 + i;
        brg[i] = (gk < K && (nb + bn) < N) ? B[(size_t)gk * N + (nb + bn)] : 0.0f;
      }
    }
  };

  load_tiles(0);  // prologue

  for (int kk = 0; kk < K; kk += 32) {
    // commit prefetched tiles to LDS (fp32 -> bf16, contiguous b128 stores)
    bf16_store8(adst0, arg[0]);
    bf16_store8(adst1, arg[1]);
    bf16_store8(bdst, brg);
    __syncthreads();

    // prefetch next K-step; global_load latency overlaps the WMMAs below
    if (kk + 32 < K) load_tiles(kk + 32);

    // fragments: one 32B LDS read each
    const v16bf a0 = *(const v16bf*)aptr0;
    const v16bf a1 = *(const v16bf*)aptr1;
    const v16bf b0 = *(const v16bf*)bptr0;
    const v16bf b1 = *(const v16bf*)bptr1;

    c00 = __builtin_amdgcn_wmma_f32_16x16x32_bf16(false, a0, false, b0,
                                                  (short)0, c00, false, false);
    c01 = __builtin_amdgcn_wmma_f32_16x16x32_bf16(false, a0, false, b1,
                                                  (short)0, c01, false, false);
    c10 = __builtin_amdgcn_wmma_f32_16x16x32_bf16(false, a1, false, b0,
                                                  (short)0, c10, false, false);
    c11 = __builtin_amdgcn_wmma_f32_16x16x32_bf16(false, a1, false, b1,
                                                  (short)0, c11, false, false);
    __syncthreads();
  }

  // ---- epilogue ----
  // C/D layout: elem r -> row = tilebase + r + (lane<16?0:8), col = lane&15
  const int rA = mb + wr + ((lane < 16) ? 0 : 8);
  const int cA = nb + wc + row16;
  const bool fullC = (mb + 128 <= M) && (nb + 64 <= N);
  if (fullC) {
    if (R) {
      #pragma unroll
      for (int r = 0; r < 8; ++r) {
        const size_t o0 = (size_t)(rA + r) * N + cA;
        const size_t o1 = (size_t)(rA + 16 + r) * N + cA;
        C[o0]      = c00[r] + R[o0];
        C[o0 + 16] = c01[r] + R[o0 + 16];
        C[o1]      = c10[r] + R[o1];
        C[o1 + 16] = c11[r] + R[o1 + 16];
      }
    } else {
      #pragma unroll
      for (int r = 0; r < 8; ++r) {
        const size_t o0 = (size_t)(rA + r) * N + cA;
        const size_t o1 = (size_t)(rA + 16 + r) * N + cA;
        C[o0]      = c00[r];
        C[o0 + 16] = c01[r];
        C[o1]      = c10[r];
        C[o1 + 16] = c11[r];
      }
    }
  } else {
    #pragma unroll
    for (int r = 0; r < 8; ++r) {
      const int g0 = rA + r, g1 = rA + 16 + r;
      if (g0 < M) {
        if (cA < N) {
          float v = c00[r];
          if (R) v += R[(size_t)g0 * N + cA];
          C[(size_t)g0 * N + cA] = v;
        }
        if (cA + 16 < N) {
          float v = c01[r];
          if (R) v += R[(size_t)g0 * N + cA + 16];
          C[(size_t)g0 * N + cA + 16] = v;
        }
      }
      if (g1 < M) {
        if (cA < N) {
          float v = c10[r];
          if (R) v += R[(size_t)g1 * N + cA];
          C[(size_t)g1 * N + cA] = v;
        }
        if (cA + 16 < N) {
          float v = c11[r];
          if (R) v += R[(size_t)g1 * N + cA + 16];
          C[(size_t)g1 * N + cA + 16] = v;
        }
      }
    }
  }
}

// ---------------------------------------------------------------------------
// RMSNorm: one block (256 thr) per row. out = w * x * rsqrt(mean(x^2)+eps)
// ---------------------------------------------------------------------------
__global__ __launch_bounds__(256)
void rmsnorm_k(const float* __restrict__ x, const float* __restrict__ w,
               float* __restrict__ out, int cols) {
  __shared__ float red[256];
  const int row = blockIdx.x;
  const float* xr = x + (size_t)row * cols;
  float s = 0.f;
  for (int c = threadIdx.x; c < cols; c += 256) { float v = xr[c]; s += v * v; }
  red[threadIdx.x] = s;
  __syncthreads();
  for (int off = 128; off > 0; off >>= 1) {
    if (threadIdx.x < off) red[threadIdx.x] += red[threadIdx.x + off];
    __syncthreads();
  }
  const float inv = rsqrtf(red[0] / (float)cols + 1e-5f);
  for (int c = threadIdx.x; c < cols; c += 256)
    out[(size_t)row * cols + c] = w[c] * xr[c] * inv;
}

// ---------------------------------------------------------------------------
// RoPE in-place: grid (rows, heads), block 32 (one wave = 32 rotation pairs)
// x layout: [row][head*64 + d], ld = row stride in floats
// ---------------------------------------------------------------------------
__global__ __launch_bounds__(32)
void rope_k(float* __restrict__ x, const int* __restrict__ pos, int ld) {
  const int row = blockIdx.x;
  const int h = blockIdx.y;
  const int j = threadIdx.x;  // 0..31
  const float p = (float)pos[row];
  const float ts = __powf(10000.0f, (float)j * (1.0f / 32.0f));
  const float rad = p / ts;
  const float s = __sinf(rad), c = __cosf(rad);
  float* xp = x + (size_t)row * ld + h * 64;
  const float x1 = xp[j], x2 = xp[j + 32];
  xp[j]      = x1 * c - x2 * s;
  xp[j + 32] = x2 * c + x1 * s;
}

// ---------------------------------------------------------------------------
// Eager attention: grid (nq, NH), block 64. GQA group = 3 (15 q / 5 kv heads).
// mode 0: dense; mode 1: valid iff (k < PL) || (k <= qi+qoff). scale=HD^-0.5.
// ---------------------------------------------------------------------------
__global__ __launch_bounds__(64)
void attn_k(const float* __restrict__ Q, const float* __restrict__ K,
            const float* __restrict__ V, float* __restrict__ O,
            int S, int qld, int kld, int old_, int mode, int qoff) {
  __shared__ float sc[SEQ];
  __shared__ float red[64];
  __shared__ float qs[64];
  const int qi = blockIdx.x;
  const int h = blockIdx.y;
  const int kvh = h / 3;
  const int t = threadIdx.x;

  qs[t] = Q[(size_t)qi * qld + h * 64 + t];
  __syncthreads();

  float lmax = -3.0e38f;
  for (int k = t; k < S; k += 64) {
    const float* kp = K + (size_t)k * kld + kvh * 64;
    float d = 0.f;
    #pragma unroll 8
    for (int i = 0; i < 64; ++i) d += qs[i] * kp[i];
    d *= 0.125f;  // 64^-0.5
    if (mode == 1 && !((k < PL) || (k <= qi + qoff))) d = -3.0e38f;
    sc[k] = d;
    lmax = fmaxf(lmax, d);
  }
  red[t] = lmax;
  __syncthreads();
  for (int off = 32; off > 0; off >>= 1) {
    if (t < off) red[t] = fmaxf(red[t], red[t + off]);
    __syncthreads();
  }
  const float m = red[0];
  __syncthreads();

  float lsum = 0.f;
  for (int k = t; k < S; k += 64) {
    float e = __expf(sc[k] - m);
    sc[k] = e;
    lsum += e;
  }
  red[t] = lsum;
  __syncthreads();
  for (int off = 32; off > 0; off >>= 1) {
    if (t < off) red[t] += red[t + off];
    __syncthreads();
  }
  const float inv = 1.0f / red[0];
  __syncthreads();

  float acc = 0.f;
  for (int k = 0; k < S; ++k)
    acc += sc[k] * V[(size_t)k * kld + kvh * 64 + t];
  O[(size_t)qi * old_ + h * 64 + t] = acc * inv;
}

// ---------------------------------------------------------------------------
// g = silu(g) * u, elementwise
// ---------------------------------------------------------------------------
__global__ __launch_bounds__(256)
void silumul_k(float* __restrict__ g, const float* __restrict__ u, int n) {
  int i = blockIdx.x * 256 + threadIdx.x;
  if (i < n) {
    float x = g[i];
    g[i] = (x / (1.0f + __expf(-x))) * u[i];
  }
}

// ---------------------------------------------------------------------------
// Host orchestration
// ---------------------------------------------------------------------------
static inline void gemm(const float* A, const float* B, float* C, const float* R,
                        int M, int N, int K, hipStream_t st) {
  dim3 g((N + 63) / 64, (M + 127) / 128);
  gemm_bf16_wmma<<<g, 256, 0, st>>>(A, B, C, R, M, N, K);
}

extern "C" void kernel_launch(void* const* d_in, const int* in_sizes, int n_in,
                              void* d_out, int out_size, void* d_ws, size_t ws_size,
                              hipStream_t stream) {
  const float* vlm_embeds = (const float*)d_in[0];
  const float* exp_embeds = (const float*)d_in[1];
  const float* vlm_q   = (const float*)d_in[2];
  const float* vlm_k   = (const float*)d_in[3];
  const float* vlm_v   = (const float*)d_in[4];
  const float* vlm_o   = (const float*)d_in[5];
  const float* vlm_ln1 = (const float*)d_in[6];
  const float* vlm_ln2 = (const float*)d_in[7];
  const float* vlm_gate = (const float*)d_in[8];
  const float* vlm_up   = (const float*)d_in[9];
  const float* vlm_down = (const float*)d_in[10];
  const float* vlm_fnorm = (const float*)d_in[11];
  const float* exp_q      = (const float*)d_in[12];
  const float* exp_k_self = (const float*)d_in[13];
  const float* exp_v_self = (const float*)d_in[14];
  const float* exp_k_cross = (const float*)d_in[15];
  const float* exp_v_cross = (const float*)d_in[16];
  const float* exp_o   = (const float*)d_in[17];
  const float* exp_ln1 = (const float*)d_in[18];
  const float* exp_ln2 = (const float*)d_in[19];
  const float* exp_gate = (const float*)d_in[20];
  const float* exp_up   = (const float*)d_in[21];
  const float* exp_down = (const float*)d_in[22];
  const float* exp_fnorm = (const float*)d_in[23];
  const int*   pos = (const int*)d_in[24];
  // d_in[25] = attention_mask: mask is computed analytically instead.

  // Workspace carve-out (fp32)
  float* ws = (float*)d_ws;
  float* vlm_st = ws;                ws += (size_t)PL * TH;   // 512x960
  float* exp_st = ws;                ws += (size_t)EL * EH;   // 128x720
  float* hv     = ws;                ws += (size_t)PL * TH;
  float* he     = ws;                ws += (size_t)EL * EH;
  float* qb     = ws;                ws += (size_t)SEQ * DQ;  // 640x960
  float* kb     = ws;                ws += (size_t)SEQ * DKV; // 640x320
  float* vb     = ws;                ws += (size_t)SEQ * DKV;
  float* keb    = ws;                ws += (size_t)PL * DKV;  // cross K
  float* veb    = ws;                ws += (size_t)PL * DKV;  // cross V
  float* attb   = ws;                ws += (size_t)SEQ * DQ;  // 640x960
  float* gV     = ws;                ws += (size_t)PL * TI;   // 512x2560
  float* uV     = ws;                ws += (size_t)PL * TI;
  float* gE     = ws;                ws += (size_t)EL * EI;   // 128x2048
  float* uE     = ws;                ws += (size_t)EL * EI;

  hipMemcpyAsync(vlm_st, vlm_embeds, (size_t)PL * TH * sizeof(float),
                 hipMemcpyDeviceToDevice, stream);
  hipMemcpyAsync(exp_st, exp_embeds, (size_t)EL * EH * sizeof(float),
                 hipMemcpyDeviceToDevice, stream);

  for (int li = 0; li < L_LAYERS; ++li) {
    // Pre-attention norms
    rmsnorm_k<<<PL, 256, 0, stream>>>(vlm_st, vlm_ln1 + (size_t)li * TH, hv, TH);
    rmsnorm_k<<<EL, 256, 0, stream>>>(exp_st, exp_ln1 + (size_t)li * EH, he, EH);

    // Q projections (concat by row offset: rows 0..511 vlm, 512..639 expert)
    gemm(hv, vlm_q + (size_t)li * TH * DQ, qb, nullptr, PL, DQ, TH, stream);
    gemm(he, exp_q + (size_t)li * EH * DQ, qb + (size_t)PL * DQ, nullptr, EL, DQ, EH, stream);
    // vlm K/V
    gemm(hv, vlm_k + (size_t)li * TH * DKV, kb, nullptr, PL, DKV, TH, stream);
    gemm(hv, vlm_v + (size_t)li * TH * DKV, vb, nullptr, PL, DKV, TH, stream);

    if ((li & 1) == 0) {
      // self K/V for expert rows
      gemm(he, exp_k_self + (size_t)(li / 2) * EH * DKV, kb + (size_t)PL * DKV,
           nullptr, EL, DKV, EH, stream);
      gemm(he, exp_v_self + (size_t)(li / 2) * EH * DKV, vb + (size_t)PL * DKV,
           nullptr, EL, DKV, EH, stream);
      // RoPE over full concat (positions align with buffer rows)
      rope_k<<<dim3(SEQ, NH), 32, 0, stream>>>(qb, pos, DQ);
      rope_k<<<dim3(SEQ, NKV), 32, 0, stream>>>(kb, pos, DKV);
      // joint attention with prefix-causal mask
      attn_k<<<dim3(SEQ, NH), 64, 0, stream>>>(qb, kb, vb, attb,
                                               SEQ, DQ, DKV, DQ, 1, 0);
    } else {
      rope_k<<<dim3(SEQ, NH), 32, 0, stream>>>(qb, pos, DQ);     // qv + qe
      rope_k<<<dim3(PL, NKV), 32, 0, stream>>>(kb, pos, DKV);    // kvr only
      // cross K/V from roped vlm K and raw vlm V
      gemm(kb, exp_k_cross + (size_t)(li / 2) * DKV * DKV, keb, nullptr,
           PL, DKV, DKV, stream);
      gemm(vb, exp_v_cross + (size_t)(li / 2) * DKV * DKV, veb, nullptr,
           PL, DKV, DKV, stream);
      // vlm self attention (dense: all j<PL valid)
      attn_k<<<dim3(PL, NH), 64, 0, stream>>>(qb, kb, vb, attb,
                                              PL, DQ, DKV, DQ, 0, 0);
      // expert cross attention over vlm keys (dense)
      attn_k<<<dim3(EL, NH), 64, 0, stream>>>(qb + (size_t)PL * DQ, keb, veb,
                                              attb + (size_t)PL * DQ,
                                              PL, DQ, DKV, DQ, 0, 0);
    }

    // Output projections with fused residual: h += att @ Wo
    gemm(attb, vlm_o + (size_t)li * DQ * TH, vlm_st, vlm_st, PL, TH, DQ, stream);
    gemm(attb + (size_t)PL * DQ, exp_o + (size_t)li * DQ * EH, exp_st, exp_st,
         EL, EH, DQ, stream);

    // MLP
    rmsnorm_k<<<PL, 256, 0, stream>>>(vlm_st, vlm_ln2 + (size_t)li * TH, hv, TH);
    rmsnorm_k<<<EL, 256, 0, stream>>>(exp_st, exp_ln2 + (size_t)li * EH, he, EH);

    gemm(hv, vlm_gate + (size_t)li * TH * TI, gV, nullptr, PL, TI, TH, stream);
    gemm(hv, vlm_up   + (size_t)li * TH * TI, uV, nullptr, PL, TI, TH, stream);
    silumul_k<<<(PL * TI + 255) / 256, 256, 0, stream>>>(gV, uV, PL * TI);
    gemm(gV, vlm_down + (size_t)li * TI * TH, vlm_st, vlm_st, PL, TH, TI, stream);

    gemm(he, exp_gate + (size_t)li * EH * EI, gE, nullptr, EL, EI, EH, stream);
    gemm(he, exp_up   + (size_t)li * EH * EI, uE, nullptr, EL, EI, EH, stream);
    silumul_k<<<(EL * EI + 255) / 256, 256, 0, stream>>>(gE, uE, EL * EI);
    gemm(gE, exp_down + (size_t)li * EI * EH, exp_st, exp_st, EL, EH, EI, stream);
  }

  // Final norms straight into d_out (vlm flat, then expert flat)
  float* out = (float*)d_out;
  rmsnorm_k<<<PL, 256, 0, stream>>>(vlm_st, vlm_fnorm, out, TH);
  rmsnorm_k<<<EL, 256, 0, stream>>>(exp_st, exp_fnorm, out + (size_t)PL * TH, EH);
}
